// CrossViewAttention_72868415144145
// MI455X (gfx1250) — matrix operand order repaired and checked by reading, not compiled, not run on hardware
//
#include <hip/hip_runtime.h>
#include <hip/hip_bf16.h>
#include <math.h>

typedef __attribute__((ext_vector_type(16))) _Float16 v16h;
typedef __attribute__((ext_vector_type(8)))  _Float16 v8h;
typedef __attribute__((ext_vector_type(8)))  float    v8f;

#define B_     4
#define NVIEW  6
#define D_     128
#define Q_     1024
#define KV     640
#define HEADS_ 4
#define DH     64
#define MD     256
#define NK     3840   // NVIEW*KV

// ---------------- wave reductions (wave32) ----------------
__device__ __forceinline__ float redmax16(float v) {
    v = fmaxf(v, __shfl_xor(v, 1));
    v = fmaxf(v, __shfl_xor(v, 2));
    v = fmaxf(v, __shfl_xor(v, 4));
    v = fmaxf(v, __shfl_xor(v, 8));
    return v;
}
__device__ __forceinline__ float redsum16(float v) {
    v += __shfl_xor(v, 1);
    v += __shfl_xor(v, 2);
    v += __shfl_xor(v, 4);
    v += __shfl_xor(v, 8);
    return v;
}
__device__ __forceinline__ float redsum32(float v) {
    v += __shfl_xor(v, 1);
    v += __shfl_xor(v, 2);
    v += __shfl_xor(v, 4);
    v += __shfl_xor(v, 8);
    v += __shfl_xor(v, 16);
    return v;
}

// ---------------- f32 weight -> f16 in WMMA-B fragment-major layout ----------
// Wp[((k0/32)*(N/16) + tn)*512 + lane*16 + e] = W[(k0 + (lane>>4)*16 + e)*N + tn*16 + (lane&15)]
// so a B fragment for (k-chunk, col-tile) is one contiguous 32B load per lane.
__global__ void cvt_perm_kernel(const float* __restrict__ W,
                                _Float16* __restrict__ Wp, int N, int K) {
    const int i = blockIdx.x * blockDim.x + threadIdx.x;
    if (i >= N * K) return;
    const int within = i & 511;          // lane*16 + e
    const int tile   = i >> 9;           // kc*(N/16) + tn
    const int ntiles = N >> 4;
    const int kc = tile / ntiles, tn = tile - kc * ntiles;
    const int lane = within >> 4, e = within & 15;
    const int hf = lane >> 4, l = lane & 15;
    const int krow = kc * 32 + hf * 16 + e;
    const int col  = tn * 16 + l;
    Wp[i] = (_Float16)W[(size_t)krow * N + col];
}

// ---------------- strided LayerNorm -> contiguous f16 rows ----------------
__global__ __launch_bounds__(256)
void ln_rows_kernel(const float* __restrict__ x, const float* __restrict__ g,
                    const float* __restrict__ bta, _Float16* __restrict__ out,
                    int rows, int inner) {
    const int wave = threadIdx.x >> 5;
    const int lane = threadIdx.x & 31;
    const int r = blockIdx.x * 8 + wave;
    if (r >= rows) return;
    const int bn = r / inner, pos = r % inner;
    const float* base = x + (size_t)bn * D_ * inner + pos;
    float v[4];
    float s = 0.f;
#pragma unroll
    for (int j = 0; j < 4; ++j) {
        v[j] = base[(size_t)(lane + 32 * j) * inner];
        s += v[j];
    }
    s = redsum32(s);
    const float mean = s * (1.f / 128.f);
    float vs = 0.f;
#pragma unroll
    for (int j = 0; j < 4; ++j) { float d = v[j] - mean; vs += d * d; }
    vs = redsum32(vs);
    const float rstd = rsqrtf(vs * (1.f / 128.f) + 1e-5f);
#pragma unroll
    for (int j = 0; j < 4; ++j) {
        const int d = lane + 32 * j;
        out[(size_t)r * D_ + d] = (_Float16)((v[j] - mean) * rstd * g[d] + bta[d]);
    }
}

// ---------------- WMMA GEMM: one wave computes a 16x64 tile ----------------
// A[MxK] row-major f16; Wperm fragment-major (see cvt_perm_kernel).
template <bool OUTH>
__global__ __launch_bounds__(32)
void gemm_wmma_kernel(const _Float16* __restrict__ A, const _Float16* __restrict__ Wperm,
                      const float* __restrict__ bias,
                      _Float16* __restrict__ outH, float* __restrict__ outF,
                      int N, int K) {
    const int tn4 = blockIdx.x;           // group of four 16-col tiles
    const int tm  = blockIdx.y;
    const int lane = threadIdx.x;
    const int hf = lane >> 4;
    const int l  = lane & 15;
    const int ntiles = N >> 4;
    v8f c[4];
    v8f zero = {};
#pragma unroll
    for (int t = 0; t < 4; ++t) c[t] = zero;

    const _Float16* arow = A + (size_t)(tm * 16 + l) * K;
    for (int k0 = 0; k0 < K; k0 += 32) {
        // A fragment: half hf -> K = hf*16 + i
        v8h alo = *(const v8h*)(arow + k0 + hf * 16);
        v8h ahi = *(const v8h*)(arow + k0 + hf * 16 + 8);
        v16h a;
#pragma unroll
        for (int i = 0; i < 8; ++i) { a[i] = alo[i]; a[i + 8] = ahi[i]; }
        const int kc = k0 >> 5;
#pragma unroll
        for (int t = 0; t < 4; ++t) {
            const int tn = tn4 * 4 + t;
            const v16h bfrag =
                *(const v16h*)(Wperm + ((size_t)(kc * ntiles + tn) << 9) + lane * 16);
            c[t] = __builtin_amdgcn_wmma_f32_16x16x32_f16(false, a, false, bfrag,
                                                          (short)0, c[t], false, false);
        }
    }
#pragma unroll
    for (int t = 0; t < 4; ++t) {
        const int col = (tn4 * 4 + t) * 16 + l;
        const float bv = bias ? bias[col] : 0.f;
#pragma unroll
        for (int r = 0; r < 8; ++r) {
            const int row = tm * 16 + r + hf * 8;   // C layout: M = vgpr + 8*half
            const float val = c[t][r] + bv;
            if (OUTH) outH[(size_t)row * N + col] = (_Float16)val;
            else      outF[(size_t)row * N + col] = val;
        }
    }
}

// ---------------- V transpose: vhh (B, n*K+key, M*Dh) -> vT (B,M,dim, n*K+key) --
__global__ void vtrans_kernel(const _Float16* __restrict__ vhh,
                              _Float16* __restrict__ vT) {
    const int i = blockIdx.x * blockDim.x + threadIdx.x;   // < B_*MD*NK
    const int nk = i % NK;
    const int rest = i / NK;          // b*256 + (m*64+d)
    const int b = rest >> 8;
    const int md = rest & 255;
    vT[i] = vhh[((size_t)b * NK + nk) * MD + md];
}

// ---------------- flash attention over concatenated views ----------------
// one wave per (b, head, 16-query tile); online softmax over 6*640 keys
__global__ __launch_bounds__(32)
void attn_kernel(const _Float16* __restrict__ qh, const _Float16* __restrict__ kh,
                 const _Float16* __restrict__ vT, const unsigned char* __restrict__ mask,
                 _Float16* __restrict__ aout) {
    __shared__ _Float16 pLDS[16 * 32];
    const int qt = blockIdx.x & 63;
    const int bm = blockIdx.x >> 6;
    const int b = bm >> 2, m = bm & 3;
    const int lane = threadIdx.x;
    const int hf = lane >> 4;
    const int l  = lane & 15;

    float mrow[8], lrow[8];
    v8f acc[4];
    v8f zero = {};
#pragma unroll
    for (int r = 0; r < 8; ++r) { mrow[r] = -1e30f; lrow[r] = 0.f; }
#pragma unroll
    for (int t = 0; t < 4; ++t) acc[t] = zero;

    for (int n = 0; n < NVIEW; ++n) {
        // Q tile A-fragments for this view (Dh=64 -> two K=32 chunks)
        const _Float16* qp =
            qh + ((size_t)((b * NVIEW + n) * Q_ + qt * 16 + l)) * MD + m * DH;
        v16h aq0, aq1;
        {
            v8h t0 = *(const v8h*)(qp + hf * 16);
            v8h t1 = *(const v8h*)(qp + 8 + hf * 16);
            v8h t2 = *(const v8h*)(qp + 32 + hf * 16);
            v8h t3 = *(const v8h*)(qp + 40 + hf * 16);
#pragma unroll
            for (int i = 0; i < 8; ++i) {
                aq0[i] = t0[i]; aq0[i + 8] = t1[i];
                aq1[i] = t2[i]; aq1[i + 8] = t3[i];
            }
        }
        // per-row visibility of this view (mask is (H,W,N) bool, 1 byte/elem)
        float vis[8];
#pragma unroll
        for (int r = 0; r < 8; ++r)
            vis[r] = mask[(size_t)(qt * 16 + r + hf * 8) * NVIEW + n] ? 1.f : 0.f;

        const _Float16* vnb = vT + ((size_t)((b * HEADS_ + m) * DH)) * NK + n * KV;

        for (int kb = 0; kb < KV; kb += 32) {
            // prefetch next key block's K rows (global_prefetch_b8)
            if (kb + 32 < KV) {
                const _Float16* pf =
                    kh + ((size_t)((b * NVIEW + n) * KV + kb + 32 + l)) * MD + m * DH;
                __builtin_prefetch((const void*)pf, 0, 0);
            }
            // ---- scores for 32 keys: two 16-key C tiles ----
            v8f s0, s1;
#pragma unroll
            for (int t = 0; t < 2; ++t) {
                const int key = kb + t * 16 + l;
                const _Float16* kp =
                    kh + ((size_t)((b * NVIEW + n) * KV + key)) * MD + m * DH + hf * 16;
                v16h bk0 = *(const v16h*)kp;         // Dh 0..31 chunk
                v16h bk1 = *(const v16h*)(kp + 32);  // Dh 32..63 chunk
                v8f cc = zero;
                cc = __builtin_amdgcn_wmma_f32_16x16x32_f16(false, aq0, false, bk0,
                                                            (short)0, cc, false, false);
                cc = __builtin_amdgcn_wmma_f32_16x16x32_f16(false, aq1, false, bk1,
                                                            (short)0, cc, false, false);
                if (t == 0) s0 = cc; else s1 = cc;
            }
            // ---- online softmax update (row stats via 16-lane shfl trees) ----
            float alpha[8];
#pragma unroll
            for (int r = 0; r < 8; ++r) {
                float a0 = vis[r] ? s0[r] * 0.125f : -1e30f;
                float a1 = vis[r] ? s1[r] * 0.125f : -1e30f;
                const float bmax = redmax16(fmaxf(a0, a1));
                const float mnew = fmaxf(mrow[r], bmax);
                alpha[r] = __expf(mrow[r] - mnew);
                const float p0 = __expf(a0 - mnew);
                const float p1 = __expf(a1 - mnew);
                pLDS[(r + hf * 8) * 32 + l]      = (_Float16)p0;
                pLDS[(r + hf * 8) * 32 + 16 + l] = (_Float16)p1;
                lrow[r] = lrow[r] * alpha[r] + redsum16(p0 + p1);
                mrow[r] = mnew;
            }
#pragma unroll
            for (int t = 0; t < 4; ++t)
#pragma unroll
                for (int r = 0; r < 8; ++r) acc[t][r] *= alpha[r];
            __syncthreads();
            // ---- P A-fragment from LDS (transposed layout) ----
            v16h pa;
            {
                const _Float16* pp = &pLDS[l * 32 + hf * 16];
                v8h p0 = *(const v8h*)pp;
                v8h p1 = *(const v8h*)(pp + 8);
#pragma unroll
                for (int i = 0; i < 8; ++i) { pa[i] = p0[i]; pa[i + 8] = p1[i]; }
            }
            // ---- O += P @ V : V B-fragments are contiguous v16h from vT ----
#pragma unroll
            for (int t = 0; t < 4; ++t) {
                const v16h vb =
                    *(const v16h*)(vnb + (size_t)(t * 16 + l) * NK + kb + hf * 16);
                acc[t] = __builtin_amdgcn_wmma_f32_16x16x32_f16(false, pa, false, vb,
                                                                (short)0, acc[t],
                                                                false, false);
            }
            __syncthreads();
        }
    }
    // ---- normalize and store f16 (B, Q, M*Dh) ----
#pragma unroll
    for (int t = 0; t < 4; ++t)
#pragma unroll
        for (int r = 0; r < 8; ++r) {
            const int q = qt * 16 + r + hf * 8;
            aout[((size_t)(b * Q_ + q)) * MD + m * DH + t * 16 + l] =
                (_Float16)(acc[t][r] / lrow[r]);
        }
}

// ---------------- z = LN_pre(z0 + bp + skip); dual f32/f16 output ----------------
__global__ __launch_bounds__(256)
void add_skip_ln_kernel(const float* __restrict__ z0, const float* __restrict__ bp,
                        const float* __restrict__ skip, const float* __restrict__ g,
                        const float* __restrict__ bt, float* __restrict__ zf,
                        _Float16* __restrict__ zh) {
    const int wave = threadIdx.x >> 5;
    const int lane = threadIdx.x & 31;
    const int r = blockIdx.x * 8 + wave;        // r in [0, 4096)
    const int b = r >> 10, q = r & 1023;
    float v[4];
    float s = 0.f;
#pragma unroll
    for (int j = 0; j < 4; ++j) {
        const int d = lane + 32 * j;
        v[j] = z0[(size_t)r * D_ + d] + bp[d] +
               skip[(size_t)b * D_ * Q_ + (size_t)d * Q_ + q];
        s += v[j];
    }
    s = redsum32(s);
    const float mean = s * (1.f / 128.f);
    float vs = 0.f;
#pragma unroll
    for (int j = 0; j < 4; ++j) { float d = v[j] - mean; vs += d * d; }
    vs = redsum32(vs);
    const float rstd = rsqrtf(vs * (1.f / 128.f) + 1e-5f);
#pragma unroll
    for (int j = 0; j < 4; ++j) {
        const int d = lane + 32 * j;
        const float y = (v[j] - mean) * rstd * g[d] + bt[d];
        zf[(size_t)r * D_ + d] = y;
        zh[(size_t)r * D_ + d] = (_Float16)y;
    }
}

// ---------------- exact GELU(h1 + b1) -> f16 ----------------
__global__ void gelu_kernel(const float* __restrict__ h1, const float* __restrict__ b1,
                            _Float16* __restrict__ g16, int n, int cols) {
    int i = blockIdx.x * blockDim.x + threadIdx.x;
    if (i >= n) return;
    const float x = h1[i] + b1[i % cols];
    const float y = 0.5f * x * (1.f + erff(x * 0.70710678118654752f));
    g16[i] = (_Float16)y;
}

// ---------------- out = LN_post(zpre + h2 + b2), transposed store (B,D,H,W) ----
__global__ __launch_bounds__(256)
void final_ln_kernel(const float* __restrict__ zpre, const float* __restrict__ h2,
                     const float* __restrict__ b2, const float* __restrict__ g,
                     const float* __restrict__ bt, float* __restrict__ out) {
    const int wave = threadIdx.x >> 5;
    const int lane = threadIdx.x & 31;
    const int r = blockIdx.x * 8 + wave;
    const int b = r >> 10, q = r & 1023;
    float v[4];
    float s = 0.f;
#pragma unroll
    for (int j = 0; j < 4; ++j) {
        const int d = lane + 32 * j;
        v[j] = zpre[(size_t)r * D_ + d] + h2[(size_t)r * D_ + d] + b2[d];
        s += v[j];
    }
    s = redsum32(s);
    const float mean = s * (1.f / 128.f);
    float vs = 0.f;
#pragma unroll
    for (int j = 0; j < 4; ++j) { float d = v[j] - mean; vs += d * d; }
    vs = redsum32(vs);
    const float rstd = rsqrtf(vs * (1.f / 128.f) + 1e-5f);
#pragma unroll
    for (int j = 0; j < 4; ++j) {
        const int d = lane + 32 * j;
        out[(size_t)b * D_ * Q_ + (size_t)d * Q_ + q] =
            (v[j] - mean) * rstd * g[d] + bt[d];
    }
}

// =====================================================================
extern "C" void kernel_launch(void* const* d_in, const int* in_sizes, int n_in,
                              void* d_out, int out_size, void* d_ws, size_t ws_size,
                              hipStream_t stream) {
    (void)in_sizes; (void)n_in; (void)out_size; (void)ws_size;
    const float* q    = (const float*)d_in[0];
    const float* k    = (const float*)d_in[1];
    const float* v    = (const float*)d_in[2];
    const float* skip = (const float*)d_in[3];
    const unsigned char* mask = (const unsigned char*)d_in[4];  // jax bool -> 1 byte
    const float* lnq_g = (const float*)d_in[5];
    const float* lnq_b = (const float*)d_in[6];
    const float* lnk_g = (const float*)d_in[7];
    const float* lnk_b = (const float*)d_in[8];
    const float* lnv_g = (const float*)d_in[9];
    const float* lnv_b = (const float*)d_in[10];
    const float* Wq = (const float*)d_in[11]; const float* bq = (const float*)d_in[12];
    const float* Wk = (const float*)d_in[13]; const float* bk = (const float*)d_in[14];
    const float* Wv = (const float*)d_in[15]; const float* bv = (const float*)d_in[16];
    const float* Wp = (const float*)d_in[17]; const float* bp = (const float*)d_in[18];
    const float* pre_g = (const float*)d_in[19]; const float* pre_b = (const float*)d_in[20];
    const float* W1 = (const float*)d_in[21]; const float* b1 = (const float*)d_in[22];
    const float* W2 = (const float*)d_in[23]; const float* b2 = (const float*)d_in[24];
    const float* post_g = (const float*)d_in[25]; const float* post_b = (const float*)d_in[26];
    float* out = (float*)d_out;

    // ---- carve workspace ----
    char* ws = (char*)d_ws;
    size_t off = 0;
    auto take = [&](size_t bytes) -> char* {
        char* p = ws + off;
        off = (off + bytes + 255) & ~(size_t)255;
        return p;
    };
    const int RQ = B_ * NVIEW * Q_;   // 24576
    const int RK = B_ * NVIEW * KV;   // 15360
    const int RZ = B_ * Q_;           // 4096
    _Float16* Wq16 = (_Float16*)take((size_t)D_ * MD * 2);
    _Float16* Wk16 = (_Float16*)take((size_t)D_ * MD * 2);
    _Float16* Wv16 = (_Float16*)take((size_t)D_ * MD * 2);
    _Float16* Wp16 = (_Float16*)take((size_t)MD * D_ * 2);
    _Float16* W116 = (_Float16*)take((size_t)D_ * 2 * D_ * 2);
    _Float16* W216 = (_Float16*)take((size_t)2 * D_ * D_ * 2);
    _Float16* qn   = (_Float16*)take((size_t)RQ * D_ * 2);
    _Float16* kn   = (_Float16*)take((size_t)RK * D_ * 2);
    _Float16* vn   = (_Float16*)take((size_t)RK * D_ * 2);
    _Float16* qhh  = (_Float16*)take((size_t)RQ * MD * 2);
    _Float16* khh  = (_Float16*)take((size_t)RK * MD * 2);
    _Float16* vhh  = (_Float16*)take((size_t)RK * MD * 2);
    _Float16* vT   = (_Float16*)take((size_t)B_ * MD * NK * 2);
    _Float16* a16  = (_Float16*)take((size_t)RZ * MD * 2);
    float*    z0   = (float*)take((size_t)RZ * D_ * 4);
    float*    zpre = (float*)take((size_t)RZ * D_ * 4);
    _Float16* zpre16 = (_Float16*)take((size_t)RZ * D_ * 2);
    float*    h1   = (float*)take((size_t)RZ * 2 * D_ * 4);
    _Float16* g16  = (_Float16*)take((size_t)RZ * 2 * D_ * 2);
    float*    h2   = (float*)take((size_t)RZ * D_ * 4);

    // 1) weights -> f16, permuted to WMMA B-fragment-major layout
    const int WN = D_ * MD;  // 32768 (all six weights same size)
    cvt_perm_kernel<<<(WN + 255) / 256, 256, 0, stream>>>(Wq, Wq16, MD, D_);
    cvt_perm_kernel<<<(WN + 255) / 256, 256, 0, stream>>>(Wk, Wk16, MD, D_);
    cvt_perm_kernel<<<(WN + 255) / 256, 256, 0, stream>>>(Wv, Wv16, MD, D_);
    cvt_perm_kernel<<<(WN + 255) / 256, 256, 0, stream>>>(Wp, Wp16, D_, MD);
    cvt_perm_kernel<<<(WN + 255) / 256, 256, 0, stream>>>(W1, W116, 2 * D_, D_);
    cvt_perm_kernel<<<(WN + 255) / 256, 256, 0, stream>>>(W2, W216, D_, 2 * D_);

    // 2) LayerNorm (strided transpose read) -> contiguous f16 rows
    ln_rows_kernel<<<RQ / 8, 256, 0, stream>>>(q, lnq_g, lnq_b, qn, RQ, Q_);
    ln_rows_kernel<<<RK / 8, 256, 0, stream>>>(k, lnk_g, lnk_b, kn, RK, KV);
    ln_rows_kernel<<<RK / 8, 256, 0, stream>>>(v, lnv_g, lnv_b, vn, RK, KV);

    // 3) projections via WMMA (f16 out + bias), 16x64 tile per wave
    gemm_wmma_kernel<true><<<dim3(MD / 64, RQ / 16), 32, 0, stream>>>(
        qn, Wq16, bq, qhh, nullptr, MD, D_);
    gemm_wmma_kernel<true><<<dim3(MD / 64, RK / 16), 32, 0, stream>>>(
        kn, Wk16, bk, khh, nullptr, MD, D_);
    gemm_wmma_kernel<true><<<dim3(MD / 64, RK / 16), 32, 0, stream>>>(
        vn, Wv16, bv, vhh, nullptr, MD, D_);

    // 3b) transpose V into key-contiguous layout for contiguous B-fragments
    vtrans_kernel<<<(B_ * MD * NK) / 256, 256, 0, stream>>>(vhh, vT);

    // 4) flash attention: B*M * (Q/16) waves
    attn_kernel<<<B_ * HEADS_ * (Q_ / 16), 32, 0, stream>>>(qhh, khh, vT, mask, a16);

    // 5) out projection (f32 out, bias fused later with skip)
    gemm_wmma_kernel<false><<<dim3(D_ / 64, RZ / 16), 32, 0, stream>>>(
        a16, Wp16, nullptr, nullptr, z0, D_, MD);

    // 6) +bp +skip, LN(pre) -> f32 + f16
    add_skip_ln_kernel<<<RZ / 8, 256, 0, stream>>>(z0, bp, skip, pre_g, pre_b,
                                                   zpre, zpre16);

    // 7) FFN up, GELU, FFN down
    gemm_wmma_kernel<false><<<dim3(2 * D_ / 64, RZ / 16), 32, 0, stream>>>(
        zpre16, W116, nullptr, nullptr, h1, 2 * D_, D_);
    gelu_kernel<<<(RZ * 2 * D_ + 255) / 256, 256, 0, stream>>>(h1, b1, g16,
                                                               RZ * 2 * D_, 2 * D_);
    gemm_wmma_kernel<false><<<dim3(D_ / 64, RZ / 16), 32, 0, stream>>>(
        g16, W216, nullptr, nullptr, h2, D_, 2 * D_);

    // 8) residual + LN(post), transposed store into (B, D, H, W)
    final_ln_kernel<<<RZ / 8, 256, 0, stream>>>(zpre, h2, b2, post_g, post_b, out);
}
